// SinkhornSelfAttention_24730421690609
// MI455X (gfx1250) — matrix operand order repaired and checked by reading, not compiled, not run on hardware
//
#include <hip/hip_runtime.h>
#include <math.h>

// ---------------- problem constants (match reference) ----------------
constexpr int kB  = 4;
constexpr int kL  = 4096;
constexpr int kD  = 1024;
constexpr int kH  = 16;
constexpr int kBS = 64;          // block size
constexpr int kHD = 64;          // head dim
constexpr int kNB = 64;          // num blocks
constexpr int k3D = 3 * kD;
constexpr float kSCALE = 0.125f; // HD^-0.5
constexpr float kINVTEMP = 1.0f / 0.7f;
constexpr int kM  = kB * kL;     // 16384 rows for the big GEMMs

// ---------------- WMMA helper types ----------------
typedef __attribute__((ext_vector_type(16))) __bf16          bf16x16;
typedef __attribute__((ext_vector_type(8)))  float           f32x8;
typedef __attribute__((ext_vector_type(8)))  unsigned short  u16x8;
typedef __attribute__((ext_vector_type(4)))  int             i32x4;

union Frag16 {
    bf16x16        v;
    u16x8          h[2];
    unsigned short s[16];
};

// round-to-nearest-even fp32 -> bf16 bits
__device__ __forceinline__ unsigned short f2bf(float f) {
    union { float f; unsigned u; } c; c.f = f;
    unsigned r = c.u + 0x7FFFu + ((c.u >> 16) & 1u);
    return (unsigned short)(r >> 16);
}

__device__ __forceinline__ f32x8 wmma_bf16(const Frag16& a, const Frag16& b, f32x8 c) {
    return __builtin_amdgcn_wmma_f32_16x16x32_bf16(
        /*neg_a=*/false, a.v, /*neg_b=*/false, b.v,
        /*c_mod=*/(short)0, c, /*reuse_a=*/false, /*reuse_b=*/false);
}

// ---------------- CDNA5 async global->LDS copy (ASYNCcnt path) ----------------
#if defined(__has_builtin)
#  if __has_builtin(__builtin_amdgcn_global_load_async_to_lds_b128)
#    define USE_ASYNC_LDS 1
#  endif
#endif

__device__ __forceinline__ void copy16_g2l(const unsigned short* g, unsigned short* l) {
#ifdef USE_ASYNC_LDS
    __builtin_amdgcn_global_load_async_to_lds_b128(
        (__attribute__((address_space(1))) i32x4*)g,
        (__attribute__((address_space(3))) i32x4*)l, /*offset=*/0, /*cpol=*/0);
#else
    *(u16x8*)l = *(const u16x8*)g;
#endif
}

__device__ __forceinline__ void copy_wait() {
#ifdef USE_ASYNC_LDS
#  if __has_builtin(__builtin_amdgcn_s_wait_asynccnt)
    __builtin_amdgcn_s_wait_asynccnt(0);
#  else
    asm volatile("s_wait_asynccnt 0" ::: "memory");
#  endif
#endif
}

// =====================================================================
// Pre-pass A: fp32 -> bf16 bulk convert (n multiple of 8)
// =====================================================================
__global__ __launch_bounds__(256)
void convert_bf16_kernel(const float* __restrict__ src, unsigned short* __restrict__ dst,
                         size_t n)
{
    size_t i = ((size_t)blockIdx.x * 256 + threadIdx.x) * 8;
    if (i >= n) return;
    u16x8 o;
    #pragma unroll
    for (int e = 0; e < 8; ++e) o[e] = f2bf(src[i + e]);
    *(u16x8*)(dst + i) = o;
}

// =====================================================================
// Pre-pass B: fp32 [R,C] -> bf16 transposed [C,R]  (R,C multiples of 32)
// =====================================================================
__global__ __launch_bounds__(256)
void convert_transpose_bf16_kernel(const float* __restrict__ src,
                                   unsigned short* __restrict__ dst, int R, int C)
{
    __shared__ unsigned short tile[32][33];
    const int rb = blockIdx.y * 32, cb = blockIdx.x * 32;
    const int tx = threadIdx.x & 31, ty = threadIdx.x >> 5;   // ty: 0..7
    #pragma unroll
    for (int i = ty; i < 32; i += 8)
        tile[i][tx] = f2bf(src[(size_t)(rb + i) * C + cb + tx]);
    __syncthreads();
    #pragma unroll
    for (int i = ty; i < 32; i += 8)
        dst[(size_t)(cb + i) * R + rb + tx] = tile[tx][i];
}

// =====================================================================
// Main GEMM: C[M,N] = A[M,K] @ Bt[N,K]^T + bias
// A, Bt pre-converted bf16 (both K-major). Block tile 128x128x32,
// 8 waves (2x4), wave tile 64x32, double-buffered LDS with async
// global->LDS staging (per-thread addresses hoisted out of the k-loop).
// =====================================================================
__global__ __launch_bounds__(256)
void gemm_bf16t_wmma(const unsigned short* __restrict__ A,
                     const unsigned short* __restrict__ Bt,
                     const float* __restrict__ bias, float* __restrict__ C,
                     int M, int N, int K)
{
    constexpr int BM = 128, BN = 128, BK = 32;
    __shared__ unsigned short As[2][BM][BK];   // [m][k] bf16
    __shared__ unsigned short Bs[2][BN][BK];   // [n][k] bf16

    const int tid   = threadIdx.x;
    const int wave  = tid >> 5;
    const int lane  = tid & 31;
    const int lrow  = lane & 15;
    const int khalf = lane >> 4;
    const int wm = (wave >> 2) * 64;           // wave row offset (2 rows of waves)
    const int wn = (wave & 3)  * 32;           // wave col offset (4 cols of waves)
    const int tileM = blockIdx.y * BM;
    const int tileN = blockIdx.x * BN;

    // ---- loop-invariant staging addresses: 2 chunks (8 bf16) each for A and B ----
    const int r0 = tid >> 2,           cc0 = (tid & 3) * 8;          // chunk tid
    const int r1 = (tid + 256) >> 2,   cc1 = ((tid + 256) & 3) * 8;  // chunk tid+256
    const unsigned short* ga0 = A  + (size_t)(tileM + r0) * K + cc0;
    const unsigned short* ga1 = A  + (size_t)(tileM + r1) * K + cc1;
    const unsigned short* gb0 = Bt + (size_t)(tileN + r0) * K + cc0;
    const unsigned short* gb1 = Bt + (size_t)(tileN + r1) * K + cc1;

    auto stage = [&](int buf, int k0) {
        copy16_g2l(ga0 + k0, &As[buf][r0][cc0]);
        copy16_g2l(ga1 + k0, &As[buf][r1][cc1]);
        copy16_g2l(gb0 + k0, &Bs[buf][r0][cc0]);
        copy16_g2l(gb1 + k0, &Bs[buf][r1][cc1]);
    };

    f32x8 acc[4][2] = {};
    stage(0, 0);
    copy_wait();
    __syncthreads();

    const int nk = K / BK;
    for (int kt = 0; kt < nk; ++kt) {
        const int buf = kt & 1;
        if (kt + 1 < nk) stage(buf ^ 1, (kt + 1) * BK);  // prefetch next slice

        Frag16 af[4], bf[2];
        #pragma unroll
        for (int i = 0; i < 4; ++i) {
            const unsigned short* ap = &As[buf][wm + i * 16 + lrow][khalf * 8];
            af[i].h[0] = *(const u16x8*)ap;
            af[i].h[1] = *(const u16x8*)(ap + 16);
        }
        #pragma unroll
        for (int j = 0; j < 2; ++j) {
            const unsigned short* bp = &Bs[buf][wn + j * 16 + lrow][khalf * 8];
            bf[j].h[0] = *(const u16x8*)bp;
            bf[j].h[1] = *(const u16x8*)(bp + 16);
        }
        #pragma unroll
        for (int i = 0; i < 4; ++i)
            #pragma unroll
            for (int j = 0; j < 2; ++j)
                acc[i][j] = wmma_bf16(af[i], bf[j], acc[i][j]);

        copy_wait();
        __syncthreads();
    }

    // epilogue: D layout lane l, vgpr v -> (row 8*(l/16)+v, col l%16)
    #pragma unroll
    for (int i = 0; i < 4; ++i)
        #pragma unroll
        for (int j = 0; j < 2; ++j) {
            int col = tileN + wn + j * 16 + lrow;
            float bv = bias[col];
            #pragma unroll
            for (int v = 0; v < 8; ++v) {
                int row = tileM + wm + i * 16 + khalf * 8 + v;
                C[(size_t)row * N + col] = acc[i][j][v] + bv;
            }
        }
}

// =====================================================================
// per-(b,h,block) means of q and k  -> q_repr, k_repr [B,H,NB,HD]
// =====================================================================
__global__ __launch_bounds__(256)
void block_repr_kernel(const float* __restrict__ qkv,
                       float* __restrict__ q_repr, float* __restrict__ k_repr)
{
    int idx = blockIdx.x * 256 + threadIdx.x;        // B*H*NB*HD = 262144
    int d = idx & (kHD - 1);
    int n = (idx >> 6) & (kNB - 1);
    int h = (idx >> 12) & (kH - 1);
    int b = idx >> 16;
    size_t base = ((size_t)b * kL + n * kBS) * k3D + h * kHD + d;
    float sq = 0.f, sk = 0.f;
    for (int i = 0; i < kBS; ++i) {
        sq += qkv[base + (size_t)i * k3D];
        sk += qkv[base + (size_t)i * k3D + kD];
    }
    q_repr[idx] = sq * (1.0f / kBS);
    k_repr[idx] = sk * (1.0f / kBS);
}

// =====================================================================
// Gumbel-Sinkhorn. One workgroup per (b,h); 64x64 log-alpha in LDS
// (padded to 65 cols so the row phase is conflict-free).
// =====================================================================
__global__ __launch_bounds__(64)
void sinkhorn_kernel(const float* __restrict__ q_repr, const float* __restrict__ k_repr,
                     const float* __restrict__ gumbel, float* __restrict__ P)
{
    __shared__ float la[kNB][kNB + 1];
    const int bh = blockIdx.x;                 // b*H + h
    const int t  = threadIdx.x;                // 0..63
    const float* qr = q_repr + (size_t)bh * kNB * kHD;
    const float* kr = k_repr + (size_t)bh * kNB * kHD;
    const float* g  = gumbel + (size_t)bh * kNB * kNB;

    for (int m = 0; m < kNB; ++m) {
        float s = 0.f;
        for (int d = 0; d < kHD; ++d) s += qr[t * kHD + d] * kr[m * kHD + d];
        la[t][m] = (s * kSCALE + g[t * kNB + m]) * kINVTEMP;
    }
    __syncthreads();

    for (int it = 0; it < 7; ++it) {
        // row LSE (axis=-1): thread t owns row t
        float mx = -1e30f;
        for (int m = 0; m < kNB; ++m) mx = fmaxf(mx, la[t][m]);
        float s = 0.f;
        for (int m = 0; m < kNB; ++m) s += __expf(la[t][m] - mx);
        float lse = mx + __logf(s);
        for (int m = 0; m < kNB; ++m) la[t][m] -= lse;
        __syncthreads();
        // col LSE (axis=-2): thread t owns column t
        mx = -1e30f;
        for (int n = 0; n < kNB; ++n) mx = fmaxf(mx, la[n][t]);
        s = 0.f;
        for (int n = 0; n < kNB; ++n) s += __expf(la[n][t] - mx);
        lse = mx + __logf(s);
        for (int n = 0; n < kNB; ++n) la[n][t] -= lse;
        __syncthreads();
    }
    for (int m = 0; m < kNB; ++m)
        P[(size_t)bh * kNB * kNB + t * kNB + m] = __expf(la[t][m]);
}

// =====================================================================
// Soft permutation of K/V blocks:
//   k_sorted[bh,n,f] = sum_m P[bh,n,m] * k_flat[bh,m,f]   (f = i*HD+d)
// =====================================================================
__global__ __launch_bounds__(256)
void sortkv_kernel(const float* __restrict__ qkv, const float* __restrict__ P,
                   float* __restrict__ k_sorted, float* __restrict__ v_sorted)
{
    __shared__ float prow[kNB];
    const int n  = blockIdx.x;
    const int bh = blockIdx.y;
    const int b  = bh >> 4, h = bh & (kH - 1);
    if (threadIdx.x < kNB)
        prow[threadIdx.x] = P[((size_t)bh * kNB + n) * kNB + threadIdx.x];
    __syncthreads();

    for (int f = threadIdx.x; f < kBS * kHD; f += 256) {
        int i = f >> 6, d = f & 63;
        size_t base = ((size_t)b * kL + i) * k3D + kD + h * kHD + d;   // k of (m=0, i, d)
        float ak = 0.f, av = 0.f;
        for (int m = 0; m < kNB; ++m) {
            float p = prow[m];
            size_t off = base + (size_t)m * kBS * k3D;
            ak += p * qkv[off];          // k
            av += p * qkv[off + kD];     // v
        }
        size_t o = ((size_t)bh * kNB + n) * (kBS * kHD) + f;
        k_sorted[o] = ak;
        v_sorted[o] = av;
    }
}

// =====================================================================
// Blocked attention. One workgroup (4 waves) per (b,h,block).
//   scores = q[64x64] @ k_local[128x64]^T * scale  (WMMA bf16)
//   softmax rows (last block masks keys >= 64)
//   out    = probs[64x128] @ v_local[128x64]        (WMMA bf16)
// out written as bf16 directly in [B,L,D] layout for the final GEMM.
// =====================================================================
__global__ __launch_bounds__(128)
void attn_kernel(const float* __restrict__ qkv,
                 const float* __restrict__ k_sorted, const float* __restrict__ v_sorted,
                 unsigned short* __restrict__ attn_bf16)
{
    __shared__ unsigned short qs[64][64];     // bf16 q  [i][d]
    __shared__ unsigned short ks[128][64];    // bf16 k  [key][d]
    __shared__ unsigned short vts[64][128];   // bf16 v^T [d][key]
    __shared__ float sc[64][128];             // fp32 scores / probs

    const int n  = blockIdx.x & (kNB - 1);
    const int bh = blockIdx.x >> 6;
    const int b  = bh >> 4, h = bh & (kH - 1);
    const int next = (n + 1) & (kNB - 1);     // jnp.roll(-1)
    const int tid  = threadIdx.x;
    const int w    = tid >> 5;
    const int lane = tid & 31;
    const int lrow  = lane & 15;
    const int khalf = lane >> 4;

    // ---- stage q / k_local / v_local^T as bf16 ----
    for (int idx = tid; idx < 64 * 64; idx += 128) {
        int i = idx >> 6, d = idx & 63;
        size_t row = (size_t)b * kL + n * kBS + i;
        qs[i][d]  = f2bf(qkv[row * k3D + h * kHD + d]);
        ks[i][d]  = f2bf(qkv[row * k3D + kD + h * kHD + d]);                 // own block K
        vts[d][i] = f2bf(qkv[row * k3D + 2 * kD + h * kHD + d]);             // own block V
        size_t so = (((size_t)bh * kNB + next) * kBS + i) * kHD + d;
        ks[64 + i][d]  = f2bf(k_sorted[so]);                                 // sorted next K
        vts[d][64 + i] = f2bf(v_sorted[so]);                                 // sorted next V
    }
    __syncthreads();

    // ---- scores: wave w owns query rows [16w,16w+16), all 8 key tiles ----
    f32x8 accS[8] = {};
    for (int kk = 0; kk < 64; kk += 32) {
        Frag16 a;
        const unsigned short* ap = &qs[16 * w + lrow][kk + khalf * 8];
        a.h[0] = *(const u16x8*)ap;
        a.h[1] = *(const u16x8*)(ap + 16);
        #pragma unroll
        for (int j = 0; j < 8; ++j) {
            Frag16 bb;
            const unsigned short* bp = &ks[16 * j + lrow][kk + khalf * 8];
            bb.h[0] = *(const u16x8*)bp;
            bb.h[1] = *(const u16x8*)(bp + 16);
            accS[j] = wmma_bf16(a, bb, accS[j]);
        }
    }
    #pragma unroll
    for (int j = 0; j < 8; ++j)
        #pragma unroll
        for (int v = 0; v < 8; ++v)
            sc[16 * w + khalf * 8 + v][16 * j + lrow] = accS[j][v] * kSCALE;
    __syncthreads();

    // ---- softmax rows (threads 0..63 each own a query row) ----
    if (tid < 64) {
        const int kmax = (n == kNB - 1) ? 64 : 128;   // last block: self only
        float mx = -1e30f;
        for (int c = 0; c < kmax; ++c) mx = fmaxf(mx, sc[tid][c]);
        float s = 0.f;
        for (int c = 0; c < kmax; ++c) { float e = __expf(sc[tid][c] - mx); sc[tid][c] = e; s += e; }
        float inv = 1.0f / s;
        for (int c = 0; c < kmax; ++c) sc[tid][c] *= inv;
        for (int c = kmax; c < 128; ++c) sc[tid][c] = 0.f;
    }
    __syncthreads();

    // ---- out = probs @ v_local : wave w owns rows [16w,16w+16), 4 d-tiles ----
    f32x8 accO[4] = {};
    for (int kk = 0; kk < 128; kk += 32) {
        Frag16 a;
        const float* arow = &sc[16 * w + lrow][0];
        #pragma unroll
        for (int e = 0; e < 8; ++e) a.s[e]     = f2bf(arow[kk + khalf * 8 + e]);
        #pragma unroll
        for (int e = 0; e < 8; ++e) a.s[8 + e] = f2bf(arow[kk + 16 + khalf * 8 + e]);
        #pragma unroll
        for (int j = 0; j < 4; ++j) {
            Frag16 bb;
            const unsigned short* bp = &vts[16 * j + lrow][kk + khalf * 8];
            bb.h[0] = *(const u16x8*)bp;
            bb.h[1] = *(const u16x8*)(bp + 16);
            accO[j] = wmma_bf16(a, bb, accO[j]);
        }
    }
    #pragma unroll
    for (int j = 0; j < 4; ++j)
        #pragma unroll
        for (int v = 0; v < 8; ++v) {
            int i = 16 * w + khalf * 8 + v;             // query index
            int d = 16 * j + lrow;                      // head-dim index
            attn_bf16[((size_t)b * kL + n * kBS + i) * kD + h * kHD + d] = f2bf(accO[j][v]);
        }
}

// =====================================================================
// Host launcher
// =====================================================================
extern "C" void kernel_launch(void* const* d_in, const int* in_sizes, int n_in,
                              void* d_out, int out_size, void* d_ws, size_t ws_size,
                              hipStream_t stream) {
    const float* x      = (const float*)d_in[0];
    const float* gumbel = (const float*)d_in[1];
    const float* W_qkv  = (const float*)d_in[2];
    const float* b_qkv  = (const float*)d_in[3];
    const float* W_out  = (const float*)d_in[4];
    const float* b_out  = (const float*)d_in[5];
    float* out = (float*)d_out;

    // ---- workspace carve-up: fp32 regions first, then bf16 regions ----
    float* wf = (float*)d_ws;
    size_t o = 0;
    float* qkv      = wf + o; o += (size_t)kM * k3D;                  // 50.3M f32
    float* q_repr   = wf + o; o += (size_t)kB * kH * kNB * kHD;
    float* k_repr   = wf + o; o += (size_t)kB * kH * kNB * kHD;
    float* P        = wf + o; o += (size_t)kB * kH * kNB * kNB;
    float* k_sorted = wf + o; o += (size_t)kB * kH * kNB * kBS * kHD; // 16.8M f32
    float* v_sorted = wf + o; o += (size_t)kB * kH * kNB * kBS * kHD;

    unsigned short* wh = (unsigned short*)(wf + o);
    size_t ho = 0;
    unsigned short* x_bf16    = wh + ho; ho += (size_t)kM * kD;       // 16.8M bf16
    unsigned short* WqkvT     = wh + ho; ho += (size_t)kD * k3D;      // [3D][D]
    unsigned short* WoutT     = wh + ho; ho += (size_t)kD * kD;       // [D][D]
    unsigned short* attn_bf16 = wh + ho; ho += (size_t)kM * kD;

    // 0) one-time operand conversions (bf16, weights transposed to [N][K])
    convert_bf16_kernel<<<((size_t)kM * kD / 8 + 255) / 256, 256, 0, stream>>>(
        x, x_bf16, (size_t)kM * kD);
    convert_transpose_bf16_kernel<<<dim3(k3D / 32, kD / 32), 256, 0, stream>>>(
        W_qkv, WqkvT, kD, k3D);
    convert_transpose_bf16_kernel<<<dim3(kD / 32, kD / 32), 256, 0, stream>>>(
        W_out, WoutT, kD, kD);

    // 1) qkv = x @ W_qkv + b_qkv        [16384 x 3072]
    gemm_bf16t_wmma<<<dim3(k3D / 128, kM / 128), 256, 0, stream>>>(
        x_bf16, WqkvT, b_qkv, qkv, kM, k3D, kD);

    // 2) block representatives (means)
    block_repr_kernel<<<(kB * kH * kNB * kHD) / 256, 256, 0, stream>>>(
        qkv, q_repr, k_repr);

    // 3) Gumbel-Sinkhorn soft permutation P
    sinkhorn_kernel<<<kB * kH, 64, 0, stream>>>(q_repr, k_repr, gumbel, P);

    // 4) soft-sort K/V blocks
    sortkv_kernel<<<dim3(kNB, kB * kH), 256, 0, stream>>>(
        qkv, P, k_sorted, v_sorted);

    // 5) blocked attention (writes bf16 A-operand of final GEMM)
    attn_kernel<<<kB * kH * kNB, 128, 0, stream>>>(
        qkv, k_sorted, v_sorted, attn_bf16);

    // 6) output projection              [16384 x 1024]
    gemm_bf16t_wmma<<<dim3(kD / 128, kM / 128), 256, 0, stream>>>(
        attn_bf16, WoutT, b_out, out, kM, kD, kD);
}
